// RowAttention_2757369004752
// MI455X (gfx1250) — matrix operand order repaired and checked by reading, not compiled, not run on hardware
//
#include <hip/hip_runtime.h>

// ---------------- types ----------------
typedef _Float16 v16h __attribute__((ext_vector_type(16)));
typedef float    v8f  __attribute__((ext_vector_type(8)));

union Frag  { v16h v; uint4 u4[2]; };
union Pack8 { uint4 u; _Float16 h[8]; };

// ---------------- problem dims ----------------
#define BATCH 8
#define CH    256
#define HH    128
#define WW    128
#define QKD   64

// ---------------- LDS layout (bytes) ----------------
#define XT_LD 264   // f16 elems per row: 256 + 8 pad   (Xt[w][c], phase 1 only)
#define QT_LD 72    // 64 + 8 pad                        (Qt[w][c], Kt[v][c])
#define VT_LD 136   // 128 + 8 pad                       (Vrow[c][v], At[w][v])
#define SR_LD 132   // 128 + 4 pad (f32)                 (Scores[w][v])

#define OFF_XT 0
#define OFF_QT (128 * XT_LD * 2)                  // 67584
#define OFF_KT (OFF_QT + 128 * QT_LD * 2)         // 86016
#define OFF_VR (OFF_KT + 128 * QT_LD * 2)         // 104448
#define OFF_SR (OFF_VR + 256 * VT_LD * 2)         // 174080
#define LDS_BYTES (OFF_SR + 128 * SR_LD * 4)      // 241664  (< 320 KB)

// ================= weight pre-conversion: fp32 -> f16, stacked [384][256] =================
__global__ void cvt_weights_kernel(const float* __restrict__ wq,
                                   const float* __restrict__ wk,
                                   const float* __restrict__ wv,
                                   _Float16* __restrict__ Wcat) {
    const int row = blockIdx.x;      // 0..383
    const int col = threadIdx.x;     // 0..255
    float v;
    if      (row < 64)  v = wq[row * CH + col];
    else if (row < 128) v = wk[(row - 64) * CH + col];
    else                v = wv[(row - 128) * CH + col];
    Wcat[row * CH + col] = (_Float16)v;
}

// ================= fused row-attention kernel: one block per (b,h) =================
__global__ __launch_bounds__(256)
void row_attn_kernel(const float* __restrict__ x,
                     const _Float16* __restrict__ Wcat,
                     const float* __restrict__ bq,
                     const float* __restrict__ bk,
                     const float* __restrict__ bv,
                     const float* __restrict__ gamma,
                     float* __restrict__ out) {
    __shared__ __align__(16) unsigned char smem[LDS_BYTES];
    _Float16* Xt = (_Float16*)(smem + OFF_XT);   // [128][XT_LD]  x^T as f16
    _Float16* At = (_Float16*)(smem + OFF_XT);   // overlay: attn [128][VT_LD] (Xt dead by then)
    _Float16* Qt = (_Float16*)(smem + OFF_QT);   // [128][QT_LD]  Q^T
    _Float16* Kt = (_Float16*)(smem + OFF_KT);   // [128][QT_LD]  K^T
    _Float16* Vr = (_Float16*)(smem + OFF_VR);   // [256][VT_LD]  V row-major
    float*    Sr = (float*)   (smem + OFF_SR);   // [128][SR_LD]  scores f32

    const int tid  = threadIdx.x;
    const int wid  = tid >> 5;       // wave id 0..7
    const int lane = tid & 31;
    const int g    = lane >> 4;      // lane group (0/1)
    const int ln   = lane & 15;

    const int bidx = blockIdx.x >> 7;    // batch
    const int h    = blockIdx.x & 127;   // row

    const float gm = gamma[0];
    const float* xs = x + (size_t)bidx * CH * HH * WW + (size_t)h * WW;

    const v8f vzero = {0.f, 0.f, 0.f, 0.f, 0.f, 0.f, 0.f, 0.f};

    // ---------- phase 0: x slab -> Xt[w][c] (f16, transposed), coalesced global reads ----------
    for (int i = tid; i < CH * WW; i += 256) {
        const int c = i >> 7, w = i & 127;
        Xt[w * XT_LD + c] = (_Float16)xs[(size_t)c * (HH * WW) + w];
    }
    __syncthreads();

    // ---------- phase 1: QKV = Wcat(384x256) x X(256x128), M-tiles 3 per wave ----------
    for (int mi = 0; mi < 3; ++mi) {
        const int mt = wid * 3 + mi;
        const int m0 = mt * 16;
        v8f acc[8];
        #pragma unroll
        for (int nt = 0; nt < 8; ++nt) acc[nt] = vzero;

        for (int k0 = 0; k0 < CH; k0 += 32) {
            Frag a;  // A layout: half j in group g -> K = g*16 + j (contiguous)
            const _Float16* ar = Wcat + (m0 + ln) * CH + k0;
            a.u4[0] = *(const uint4*)(ar + g * 16);
            a.u4[1] = *(const uint4*)(ar + g * 16 + 8);
            #pragma unroll
            for (int nt = 0; nt < 8; ++nt) {
                Frag bf; // B layout: half j in group g -> K = g*16 + j (contiguous)
                const _Float16* br = Xt + (nt * 16 + ln) * XT_LD + k0 + g * 16;
                bf.u4[0] = *(const uint4*)(br);
                bf.u4[1] = *(const uint4*)(br + 8);
                acc[nt] = __builtin_amdgcn_wmma_f32_16x16x32_f16(
                    false, a.v, false, bf.v, (short)0, acc[nt], false, false);
            }
        }
        // ---------- epilogue: bias add + scatter to Qt / Kt / Vr ----------
        if (m0 < 128) {                       // Q rows (0..63) or K rows (64..127)
            const bool isQ = (m0 < 64);
            const int o0 = (isQ ? m0 : m0 - 64) + 8 * g;
            const float* bias = (isQ ? bq : bk) + o0;
            _Float16* dst = isQ ? Qt : Kt;
            float bb[8];
            #pragma unroll
            for (int r = 0; r < 8; ++r) bb[r] = bias[r];
            #pragma unroll
            for (int nt = 0; nt < 8; ++nt) {
                const int w = nt * 16 + ln;   // D: m = m0 + 8g + r, n = w
                Pack8 p;
                #pragma unroll
                for (int r = 0; r < 8; ++r) p.h[r] = (_Float16)(acc[nt][r] + bb[r]);
                *(uint4*)(dst + w * QT_LD + o0) = p.u;   // 8 consecutive f16 = 16B
            }
        } else {                              // V rows (channels 0..255)
            const int c0 = m0 - 128 + 8 * g;
            float bb[8];
            #pragma unroll
            for (int r = 0; r < 8; ++r) bb[r] = bv[c0 + r];
            #pragma unroll
            for (int nt = 0; nt < 8; ++nt) {
                const int w = nt * 16 + ln;
                #pragma unroll
                for (int r = 0; r < 8; ++r)
                    Vr[(c0 + r) * VT_LD + w] = (_Float16)(acc[nt][r] + bb[r]);
            }
        }
    }
    __syncthreads();

    // ---------- phase 2a: scores S = Q^T(128x64) x K(64x128), 1 M-tile per wave ----------
    {
        const int m0 = wid * 16;
        v8f acc[8];
        #pragma unroll
        for (int nt = 0; nt < 8; ++nt) acc[nt] = vzero;

        for (int k0 = 0; k0 < QKD; k0 += 32) {
            Frag a;
            const _Float16* ar = Qt + (m0 + ln) * QT_LD + k0;
            a.u4[0] = *(const uint4*)(ar + g * 16);
            a.u4[1] = *(const uint4*)(ar + g * 16 + 8);
            #pragma unroll
            for (int nt = 0; nt < 8; ++nt) {
                Frag bf;
                const _Float16* br = Kt + (nt * 16 + ln) * QT_LD + k0 + g * 16;
                bf.u4[0] = *(const uint4*)(br);
                bf.u4[1] = *(const uint4*)(br + 8);
                acc[nt] = __builtin_amdgcn_wmma_f32_16x16x32_f16(
                    false, a.v, false, bf.v, (short)0, acc[nt], false, false);
            }
        }
        #pragma unroll
        for (int nt = 0; nt < 8; ++nt)
            #pragma unroll
            for (int r = 0; r < 8; ++r)
                Sr[(m0 + 8 * g + r) * SR_LD + nt * 16 + ln] = acc[nt][r];
    }
    __syncthreads();

    // ---------- softmax over keys: 2 threads per row, f32, writes At (overlays dead Xt) ----------
    {
        const int row = tid >> 1, hf = tid & 1;
        const float* sp = Sr + row * SR_LD + hf * 64;
        float mx = -3.0e38f;
        for (int i = 0; i < 64; ++i) mx = fmaxf(mx, sp[i]);
        mx = fmaxf(mx, __shfl_xor(mx, 1, 32));
        _Float16* ap = At + row * VT_LD + hf * 64;
        float sum = 0.f;
        for (int i = 0; i < 64; ++i) {
            float e = __expf(sp[i] - mx);
            sum += e;
            ap[i] = (_Float16)e;   // unnormalized, in (0,1]
        }
        sum += __shfl_xor(sum, 1, 32);
        const float inv = 1.0f / sum;
        for (int i = 0; i < 64; ++i)
            ap[i] = (_Float16)((float)ap[i] * inv);
    }
    __syncthreads();

    // ---------- phase 2b: O = V(256x128) x attn^T(128x128); fuse gamma*O + x ----------
    for (int mi = 0; mi < 2; ++mi) {
        const int c0 = (wid * 2 + mi) * 16;
        v8f acc[8];
        #pragma unroll
        for (int nt = 0; nt < 8; ++nt) acc[nt] = vzero;

        for (int k0 = 0; k0 < WW; k0 += 32) {
            Frag a;
            const _Float16* ar = Vr + (c0 + ln) * VT_LD + k0;
            a.u4[0] = *(const uint4*)(ar + g * 16);
            a.u4[1] = *(const uint4*)(ar + g * 16 + 8);
            #pragma unroll
            for (int nt = 0; nt < 8; ++nt) {
                Frag bf;
                const _Float16* br = At + (nt * 16 + ln) * VT_LD + k0 + g * 16;
                bf.u4[0] = *(const uint4*)(br);
                bf.u4[1] = *(const uint4*)(br + 8);
                acc[nt] = __builtin_amdgcn_wmma_f32_16x16x32_f16(
                    false, a.v, false, bf.v, (short)0, acc[nt], false, false);
            }
        }
        #pragma unroll
        for (int nt = 0; nt < 8; ++nt) {
            const int w = nt * 16 + ln;
            #pragma unroll
            for (int r = 0; r < 8; ++r) {
                const int c = c0 + 8 * g + r;
                const size_t idx = (((size_t)bidx * CH + c) * HH + h) * WW + w;
                out[idx] = gm * acc[nt][r] + x[idx];
            }
        }
    }
}

// ================= launcher =================
extern "C" void kernel_launch(void* const* d_in, const int* in_sizes, int n_in,
                              void* d_out, int out_size, void* d_ws, size_t ws_size,
                              hipStream_t stream) {
    (void)in_sizes; (void)n_in; (void)out_size; (void)ws_size;
    const float* x     = (const float*)d_in[0];
    const float* wq    = (const float*)d_in[1];
    const float* bq    = (const float*)d_in[2];
    const float* wk    = (const float*)d_in[3];
    const float* bk    = (const float*)d_in[4];
    const float* wv    = (const float*)d_in[5];
    const float* bv    = (const float*)d_in[6];
    const float* gamma = (const float*)d_in[7];

    _Float16* Wcat = (_Float16*)d_ws;   // 384*256*2 = 196608 bytes of scratch

    cvt_weights_kernel<<<384, 256, 0, stream>>>(wq, wk, wv, Wcat);
    row_attn_kernel<<<BATCH * HH, 256, 0, stream>>>(x, Wcat, bq, bk, bv, gamma,
                                                    (float*)d_out);
}
